// SelDimeNet_47115791237974
// MI455X (gfx1250) — matrix operand order repaired and checked by reading, not compiled, not run on hardware
//
#include <hip/hip_runtime.h>

typedef __attribute__((ext_vector_type(16))) __bf16       v16bf;
typedef __attribute__((ext_vector_type(8)))  float        v8f;
typedef __attribute__((ext_vector_type(4)))  unsigned int u32x4;

// ---------------- problem constants ----------------
static constexpr int kE   = 400000;
static constexpr int kT   = 1200000;
static constexpr int kH   = 128;
static constexpr int kB   = 8;     // num_bilinear
static constexpr int kNR  = 6;
static constexpr int kNSR = 42;    // NS*NR
static constexpr int kMatElems = kH * kH;   // 16384 elements per 128x128 matrix
static constexpr int kNumMats  = 25;
static constexpr int kOrdN     = kT + 256;

// packed-matrix ids
static constexpr int M_JI = 0, M_KJ = 1, M_SEL = 2, M_W = 10,
                     M_RB0 = 18, M_RB1 = 19, M_LIN = 20, M_RA = 21;

// ---------------- workspace layout (bytes) ----------------
static constexpr size_t OFF_PACKED = 0;
static constexpr size_t OFF_XKJ    = OFF_PACKED + (size_t)kNumMats * kMatElems * 2;
static constexpr size_t OFF_MSG    = OFF_XKJ + (size_t)kE * kH * 2;
static constexpr size_t OFF_SBFT   = OFF_MSG + (size_t)kE * kH * 4;
static constexpr size_t OFF_ORDER  = OFF_SBFT + (size_t)kT * kB * 4;
static constexpr size_t OFF_META   = OFF_ORDER + (size_t)kOrdN * 4;

// ---------------- helpers ----------------
union AB16 {
  v16bf v;
  unsigned short u[16];
  u32x4 q[2];
};

__device__ __forceinline__ unsigned short f2bf(float f) {
  union { float f; unsigned int u; } c; c.f = f;
  unsigned int r = c.u + 0x7FFFu + ((c.u >> 16) & 1u);   // round-to-nearest-even
  return (unsigned short)(r >> 16);
}

__device__ __forceinline__ float silu(float x) { return x / (1.0f + __expf(-x)); }

__device__ __forceinline__ v8f v8f_zero() {
  v8f z = {0.f, 0.f, 0.f, 0.f, 0.f, 0.f, 0.f, 0.f};
  return z;
}

__device__ __forceinline__ v8f wmma_bf16(v16bf a, v16bf b, v8f c) {
  return __builtin_amdgcn_wmma_f32_16x16x32_bf16(false, a, false, b, (short)0, c,
                                                 false, false);
}

// A fragment (16x32 bf16, columns kbase..kbase+31) from an LDS [16][128] tile.
// Lane L holds row M=L%16; elems 0..7 -> K=kbase+8*hi+e, elems 8..15 -> K=kbase+16+8*hi+(e-8).
__device__ __forceinline__ v16bf load_a_lds(const unsigned short* lds, int lane, int kbase) {
  const int m  = lane & 15;
  const int hi = lane >> 4;
  AB16 f;
  f.q[0] = *(const u32x4*)(lds + m * kH + kbase + 8 * hi);
  f.q[1] = *(const u32x4*)(lds + m * kH + kbase + 16 + 8 * hi);
  return f.v;
}

// B fragment from fragment-packed global weights: 32 contiguous bytes per lane.
__device__ __forceinline__ v16bf load_b_pk(const unsigned short* pk, int ntile, int kk, int lane) {
  const unsigned short* s = pk + (size_t)((((ntile << 2) + kk) << 5) + lane) * 16;
  AB16 f;
  f.q[0] = ((const u32x4*)s)[0];
  f.q[1] = ((const u32x4*)s)[1];
  return f.v;
}

__device__ __forceinline__ int bucket_of(float ang) {
  int b = (int)(ang * (8.0f / 3.141593f));
  return b < 0 ? 0 : (b > 7 ? 7 : b);
}

// ---------------- kernels ----------------

// zero msg (b128 stores), invalidate order, zero meta
__global__ void k_init(u32x4* __restrict__ msg4, unsigned int* __restrict__ order,
                       int* __restrict__ meta) {
  const int n4 = kE * kH / 4;   // 12.8M 16B stores
  const int stride = gridDim.x * blockDim.x;
  const u32x4 z = {0u, 0u, 0u, 0u};
  for (int i = blockIdx.x * blockDim.x + threadIdx.x; i < n4; i += stride) {
    msg4[i] = z;
    if (i < kOrdN) order[i] = 0xFFFFFFFFu;
    if (i < 64) meta[i] = 0;
  }
}

// convert + transpose all 128x128 weight matrices into WMMA B-fragment order (bf16)
__global__ void k_prep(const float* __restrict__ ji_w, const float* __restrict__ kj_w,
                       const float* __restrict__ sel_w, const float* __restrict__ Wbl,
                       const float* __restrict__ rb_w, const float* __restrict__ lin_w,
                       const float* __restrict__ ra_w, unsigned short* __restrict__ packed) {
  const int m = blockIdx.x;
  for (int idx = threadIdx.x; idx < kMatElems; idx += blockDim.x) {
    const int e    = idx & 15;
    const int lane = (idx >> 4) & 31;
    const int kk   = (idx >> 9) & 3;
    const int nt   = idx >> 11;
    const int k = kk * 32 + ((lane & 16) >> 1) + (e & 7) + ((e >> 3) << 4);   // K = kbase + 8*(lane>=16) + (e&7) + 16*(e>=8)
    const int n = nt * 16 + (lane & 15);
    float v;
    if (m == M_JI)        v = ji_w[k * kH + n];
    else if (m == M_KJ)   v = kj_w[k * kH + n];
    else if (m < M_W)     v = sel_w[(m - M_SEL) * kMatElems + k * kH + n];
    else if (m < M_RB0)   v = Wbl[((size_t)n * kB + (m - M_W)) * kH + k]; // W[i,b,l] -> B[l][i]
    else if (m == M_RB0)  v = rb_w[k * kH + n];
    else if (m == M_RB1)  v = rb_w[kMatElems + k * kH + n];
    else if (m == M_LIN)  v = lin_w[k * kH + n];
    else                  v = ra_w[(m - M_RA) * kMatElems + k * kH + n];
    packed[(size_t)m * kMatElems + idx] = f2bf(v);
  }
}

// sbf_t = sbf @ lin_sbf_w  (+ per-bucket histogram of angle buckets)
__global__ __launch_bounds__(256) void k_sbf_sel(
    const float* __restrict__ sbf, const float* __restrict__ lin_sbf_w,
    const float* __restrict__ angle, float* __restrict__ sbft, int* __restrict__ meta) {
  __shared__ float lw[kNSR * kB];
  __shared__ int lcnt[8];
  for (int i = threadIdx.x; i < kNSR * kB; i += 256) lw[i] = lin_sbf_w[i];
  if (threadIdx.x < 8) lcnt[threadIdx.x] = 0;
  __syncthreads();
  const int t = blockIdx.x * 256 + threadIdx.x;
  if (t < kT) {
    float acc[kB];
#pragma unroll
    for (int b = 0; b < kB; ++b) acc[b] = 0.0f;
    const float* srow = sbf + (size_t)t * kNSR;
    for (int q = 0; q < kNSR; ++q) {
      const float s = srow[q];
#pragma unroll
      for (int b = 0; b < kB; ++b) acc[b] += s * lw[q * kB + b];
    }
#pragma unroll
    for (int b = 0; b < kB; ++b) sbft[(size_t)t * kB + b] = acc[b];
    atomicAdd(&lcnt[bucket_of(angle[t])], 1);
  }
  __syncthreads();
  if (threadIdx.x < 8) atomicAdd(&meta[threadIdx.x], lcnt[threadIdx.x]);
}

// padded (multiple-of-16) bucket bases
__global__ void k_base(int* __restrict__ meta) {
  if (threadIdx.x == 0 && blockIdx.x == 0) {
    int acc = 0;
    for (int b = 0; b < 8; ++b) {
      meta[8 + b] = acc;
      acc += ((meta[b] + 15) >> 4) << 4;
    }
    meta[16] = acc;
    for (int b = 0; b < 8; ++b) meta[24 + b] = 0;
  }
}

// counting-sort scatter of triplet ids into bucket-pure, 16-padded order[]
__global__ void k_scatter(const float* __restrict__ angle, unsigned int* __restrict__ order,
                          int* __restrict__ meta) {
  const int t = blockIdx.x * blockDim.x + threadIdx.x;
  if (t >= kT) return;
  const int b = bucket_of(angle[t]);
  const int pos = atomicAdd(&meta[24 + b], 1);
  order[meta[8 + b] + pos] = (unsigned int)t;
}

// x_kj = bf16(silu(x@Wkj+b) * (rbf@lin_rbf_w))
__global__ __launch_bounds__(256) void k_edge_front(
    const float* __restrict__ x, const float* __restrict__ rbf,
    const float* __restrict__ lin_rbf_w, const float* __restrict__ kj_b,
    const unsigned short* __restrict__ packed, unsigned short* __restrict__ xkj) {
  __shared__ alignas(16) unsigned short lx[16 * kH];
  __shared__ float lrbf[16 * kNR];
  const int tile = blockIdx.x;
  const int tid = threadIdx.x;
  for (int i = tid; i < 16 * kH; i += 256) {
    const int row = i >> 7, cc = i & 127;
    lx[i] = f2bf(x[(size_t)(tile * 16 + row) * kH + cc]);
  }
  if (tid < 16 * kNR) {
    const int row = tid / kNR, q = tid % kNR;
    lrbf[tid] = rbf[(size_t)(tile * 16 + row) * kNR + q];
  }
  __syncthreads();
  const int wave = tid >> 5, lane = tid & 31;
  const int col = wave * 16 + (lane & 15);
  const int hi = lane >> 4;
  v8f ckj = v8f_zero();
  const unsigned short* pkj = packed + (size_t)M_KJ * kMatElems;
#pragma unroll
  for (int kk = 0; kk < 4; ++kk)
    ckj = wmma_bf16(load_a_lds(lx, lane, kk * 32), load_b_pk(pkj, wave, kk, lane), ckj);
  const float bkj = kj_b[col];
  float wr[kNR];
#pragma unroll
  for (int q = 0; q < kNR; ++q) wr[q] = lin_rbf_w[q * kH + col];
#pragma unroll
  for (int r = 0; r < 8; ++r) {
    const int m = r + 8 * hi;
    float re = 0.f;
#pragma unroll
    for (int q = 0; q < kNR; ++q) re += lrbf[m * kNR + q] * wr[q];
    xkj[(size_t)(tile * 16 + m) * kH + col] = f2bf(silu(ckj[r] + bkj) * re);
  }
}

// per bucket-pure 16-triplet tile: gather -> sel GEMM -> bilinear -> atomic scatter
__global__ __launch_bounds__(256) void k_triplet(
    const unsigned short* __restrict__ xkj, const float* __restrict__ sbft,
    const unsigned int* __restrict__ order, const int* __restrict__ meta,
    const int* __restrict__ idx_kj, const int* __restrict__ idx_ji,
    const unsigned short* __restrict__ packed, float* __restrict__ msg) {
  __shared__ alignas(16) unsigned short lxk[16 * kH];
  __shared__ alignas(16) unsigned short lxk2[16 * kH];
  __shared__ float lsbf[16 * kB];
  __shared__ int ltid[16], lkj[16], lji[16];
  const int tid = threadIdx.x;
  const int tstart = blockIdx.x * 16;
  int s = 0;                                   // bucket (uniform over block)
#pragma unroll
  for (int b = 0; b < 8; ++b)
    if (tstart >= meta[8 + b] && tstart < meta[8 + b + 1]) s = b;
  if (tid < 16) {
    const int t = (int)order[tstart + tid];    // 0xFFFFFFFF pad -> -1
    ltid[tid] = t;
    lkj[tid] = (t >= 0) ? idx_kj[t] : -1;
    lji[tid] = (t >= 0) ? idx_ji[t] : -1;
  }
  __syncthreads();
  {
    const int row = tid >> 4, seg = tid & 15;  // 16B segment gather of bf16 rows
    const int e = lkj[row];
    u32x4 v = {0u, 0u, 0u, 0u};
    if (e >= 0) v = *(const u32x4*)(xkj + (size_t)e * kH + seg * 8);
    *(u32x4*)(lxk + row * kH + seg * 8) = v;
  }
  if (tid < 16 * kB) {
    const int row = tid >> 3, b = tid & 7;
    const int t = ltid[row];
    lsbf[tid] = (t >= 0) ? sbft[(size_t)t * kB + b] : 0.f;
  }
  __syncthreads();
  const int wave = tid >> 5, lane = tid & 31;
  const int col = wave * 16 + (lane & 15);
  const int hi = lane >> 4;
  {                                            // selection GEMM (uniform sel_w[s])
    const unsigned short* psel = packed + (size_t)(M_SEL + s) * kMatElems;
    v8f c = v8f_zero();
#pragma unroll
    for (int kk = 0; kk < 4; ++kk)
      c = wmma_bf16(load_a_lds(lxk, lane, kk * 32), load_b_pk(psel, wave, kk, lane), c);
#pragma unroll
    for (int r = 0; r < 8; ++r) lxk2[(r + 8 * hi) * kH + col] = f2bf(c[r]);
  }
  __syncthreads();
  v16bf af[4];                                 // keep full A resident across all 8 b
#pragma unroll
  for (int kk = 0; kk < 4; ++kk) af[kk] = load_a_lds(lxk2, lane, kk * 32);
  v8f acc = v8f_zero();
#pragma unroll
  for (int b = 0; b < kB; ++b) {
    const unsigned short* pw = packed + (size_t)(M_W + b) * kMatElems;
    v8f cb = v8f_zero();
#pragma unroll
    for (int kk = 0; kk < 4; ++kk)
      cb = wmma_bf16(af[kk], load_b_pk(pw, wave, kk, lane), cb);
#pragma unroll
    for (int r = 0; r < 8; ++r)
      acc[r] += lsbf[(r + 8 * hi) * kB + b] * cb[r];
  }
#pragma unroll
  for (int r = 0; r < 8; ++r) {
    const int m = r + 8 * hi;
    const int e = lji[m];
    if (e >= 0) atomicAdd(msg + (size_t)e * kH + col, acc[r]);
  }
}

__device__ __forceinline__ v8f gemm_stage(const unsigned short* A, const unsigned short* P,
                                          const float* bias, int wave, int lane) {
  v8f c = v8f_zero();
#pragma unroll
  for (int kk = 0; kk < 4; ++kk)
    c = wmma_bf16(load_a_lds(A, lane, kk * 32), load_b_pk(P, wave, kk, lane), c);
  const float bv = bias[wave * 16 + (lane & 15)];
#pragma unroll
  for (int r = 0; r < 8; ++r) c[r] += bv;
  return c;
}

// lin_ji GEMM + whole post-message residual chain (8 GEMMs), LDS ping-pong
__global__ __launch_bounds__(256) void k_edge_post(
    const float* __restrict__ x, const float* __restrict__ msg,
    const unsigned short* __restrict__ packed, const float* __restrict__ ji_b,
    const float* __restrict__ rb_b, const float* __restrict__ lin_b,
    const float* __restrict__ ra_b, float* __restrict__ out) {
  __shared__ alignas(16) unsigned short lbf0[16 * kH];
  __shared__ alignas(16) unsigned short lbf1[16 * kH];
  __shared__ float lh[16 * kH];
  const int tile = blockIdx.x;
  const int tid = threadIdx.x;
  for (int i = tid; i < 16 * kH; i += 256) {
    const int row = i >> 7, cc = i & 127;
    lbf0[i] = f2bf(x[(size_t)(tile * 16 + row) * kH + cc]);
  }
  __syncthreads();
  const int wave = tid >> 5, lane = tid & 31;
  const int col = wave * 16 + (lane & 15);
  const int hi = lane >> 4;
  v8f c;
  // S0: h0 = silu(x @ Wji + bji) + msg
  c = gemm_stage(lbf0, packed + (size_t)M_JI * kMatElems, ji_b, wave, lane);
#pragma unroll
  for (int r = 0; r < 8; ++r) {
    const int m = r + 8 * hi;
    const float h0 = silu(c[r]) + msg[(size_t)(tile * 16 + m) * kH + col];
    lh[m * kH + col] = h0;
    lbf1[m * kH + col] = f2bf(h0);
  }
  __syncthreads();
  // S1: u = silu(h0 @ rb0 + b00)
  c = gemm_stage(lbf1, packed + (size_t)M_RB0 * kMatElems, rb_b, wave, lane);
#pragma unroll
  for (int r = 0; r < 8; ++r) lbf0[(r + 8 * hi) * kH + col] = f2bf(silu(c[r]));
  __syncthreads();
  // S2: h1 = h0 + silu(u @ rb1 + b01)
  c = gemm_stage(lbf0, packed + (size_t)M_RB1 * kMatElems, rb_b + kH, wave, lane);
#pragma unroll
  for (int r = 0; r < 8; ++r) {
    const int m = r + 8 * hi;
    const float h1 = lh[m * kH + col] + silu(c[r]);
    lh[m * kH + col] = h1;
    lbf1[m * kH + col] = f2bf(h1);
  }
  __syncthreads();
  // S3: h2 = silu(h1 @ lin + lb) + x
  c = gemm_stage(lbf1, packed + (size_t)M_LIN * kMatElems, lin_b, wave, lane);
#pragma unroll
  for (int r = 0; r < 8; ++r) {
    const int m = r + 8 * hi;
    const float h2 = silu(c[r]) + x[(size_t)(tile * 16 + m) * kH + col];
    lh[m * kH + col] = h2;
    lbf0[m * kH + col] = f2bf(h2);
  }
  __syncthreads();
  // S4: u2 = silu(h2 @ ra00 + b)
  c = gemm_stage(lbf0, packed + (size_t)(M_RA + 0) * kMatElems, ra_b, wave, lane);
#pragma unroll
  for (int r = 0; r < 8; ++r) lbf1[(r + 8 * hi) * kH + col] = f2bf(silu(c[r]));
  __syncthreads();
  // S5: h3 = h2 + silu(u2 @ ra01 + b)
  c = gemm_stage(lbf1, packed + (size_t)(M_RA + 1) * kMatElems, ra_b + kH, wave, lane);
#pragma unroll
  for (int r = 0; r < 8; ++r) {
    const int m = r + 8 * hi;
    const float h3 = lh[m * kH + col] + silu(c[r]);
    lh[m * kH + col] = h3;
    lbf0[m * kH + col] = f2bf(h3);
  }
  __syncthreads();
  // S6: u3 = silu(h3 @ ra10 + b)
  c = gemm_stage(lbf0, packed + (size_t)(M_RA + 2) * kMatElems, ra_b + 2 * kH, wave, lane);
#pragma unroll
  for (int r = 0; r < 8; ++r) lbf1[(r + 8 * hi) * kH + col] = f2bf(silu(c[r]));
  __syncthreads();
  // S7: out = h3 + silu(u3 @ ra11 + b)
  c = gemm_stage(lbf1, packed + (size_t)(M_RA + 3) * kMatElems, ra_b + 3 * kH, wave, lane);
#pragma unroll
  for (int r = 0; r < 8; ++r) {
    const int m = r + 8 * hi;
    out[(size_t)(tile * 16 + m) * kH + col] = lh[m * kH + col] + silu(c[r]);
  }
}

// ---------------- host launcher ----------------
extern "C" void kernel_launch(void* const* d_in, const int* in_sizes, int n_in,
                              void* d_out, int out_size, void* d_ws, size_t ws_size,
                              hipStream_t stream) {
  (void)in_sizes; (void)n_in; (void)out_size; (void)ws_size;
  const float* x         = (const float*)d_in[0];
  const float* rbf       = (const float*)d_in[1];
  const float* sbf       = (const float*)d_in[2];
  const float* angle     = (const float*)d_in[3];
  const int*   idx_kj    = (const int*)d_in[4];
  const int*   idx_ji    = (const int*)d_in[5];
  const float* lin_rbf_w = (const float*)d_in[6];
  const float* lin_sbf_w = (const float*)d_in[7];
  const float* lin_kj_w  = (const float*)d_in[8];
  const float* lin_kj_b  = (const float*)d_in[9];
  const float* lin_ji_w  = (const float*)d_in[10];
  const float* lin_ji_b  = (const float*)d_in[11];
  const float* Wbl       = (const float*)d_in[12];
  const float* sel_w     = (const float*)d_in[13];
  const float* rb_w      = (const float*)d_in[14];
  const float* rb_b      = (const float*)d_in[15];
  const float* ra_w      = (const float*)d_in[16];
  const float* ra_b      = (const float*)d_in[17];
  const float* lin_w     = (const float*)d_in[18];
  const float* lin_b     = (const float*)d_in[19];

  char* ws = (char*)d_ws;
  unsigned short* packed = (unsigned short*)(ws + OFF_PACKED);
  unsigned short* xkj    = (unsigned short*)(ws + OFF_XKJ);
  float* msg             = (float*)(ws + OFF_MSG);
  float* sbft            = (float*)(ws + OFF_SBFT);
  unsigned int* order    = (unsigned int*)(ws + OFF_ORDER);
  int* meta              = (int*)(ws + OFF_META);

  k_init<<<2048, 256, 0, stream>>>((u32x4*)msg, order, meta);
  k_prep<<<kNumMats, 256, 0, stream>>>(lin_ji_w, lin_kj_w, sel_w, Wbl, rb_w, lin_w,
                                       ra_w, packed);
  k_sbf_sel<<<(kT + 255) / 256, 256, 0, stream>>>(sbf, lin_sbf_w, angle, sbft, meta);
  k_base<<<1, 32, 0, stream>>>(meta);
  k_scatter<<<(kT + 255) / 256, 256, 0, stream>>>(angle, order, meta);
  k_edge_front<<<kE / 16, 256, 0, stream>>>(x, rbf, lin_rbf_w, lin_kj_b, packed, xkj);
  k_triplet<<<kT / 16 + 8, 256, 0, stream>>>(xkj, sbft, order, meta, idx_kj, idx_ji,
                                             packed, msg);
  k_edge_post<<<kE / 16, 256, 0, stream>>>(x, msg, packed, lin_ji_b, rb_b, lin_b,
                                           ra_b, (float*)d_out);
}
